// MultiplicativeAttention_38190849196269
// MI455X (gfx1250) — matrix operand order repaired and checked by reading, not compiled, not run on hardware
//
#include <hip/hip_runtime.h>
#include <hip/hip_bf16.h>
#include <math.h>

// ---------------------------------------------------------------------------
// MultiplicativeAttention for MI455X (gfx1250, wave32, WMMA).
// All GEMMs: v_wmma_f32_16x16x32_bf16. LDS double-buffered pipelines; every
// fragment load is a contiguous 32-byte ds_load_b128 (fragment-major A,
// n-major B); W transpose staged as 4kx2n b64 patches (no 16-bit scatter).
// ---------------------------------------------------------------------------

typedef __attribute__((ext_vector_type(16))) __bf16 v16bf;
typedef __attribute__((ext_vector_type(8)))  __bf16 v8bf;
typedef __attribute__((ext_vector_type(4)))  __bf16 v4bf;
typedef __attribute__((ext_vector_type(8)))  float  v8f;
typedef __attribute__((ext_vector_type(4)))  float  v4f;
typedef __attribute__((ext_vector_type(2)))  float  v2f;

#define D_MODEL 1024
#define SEQ     2048
#define NBATCH  2
#define NHEADS  16
#define DHEAD   64
#define MTOT    (NBATCH * SEQ)   // 4096 token rows
#define EPSV    1e-6f

__device__ __forceinline__ v8f wmma_bf16(v16bf a, v16bf b, v8f c) {
  return __builtin_amdgcn_wmma_f32_16x16x32_bf16(
      false, a, false, b, (short)0, c, false, false);
}

// Fragment-major A storage. A 16x32 layout (contiguous k, matching B) is
//   lane = m | (k4<<4) ;  pos = k[3:0]
// so 8 consecutive k (c0 multiple of 8) at fixed m are 8 consecutive vector
// positions of one lane. Storage ordered [msub*nksub + ksub][lane][16].
__device__ __forceinline__ int afrag_group(int r, int c0, int nksub) {
  const int msub = r >> 4;
  const int ksub = c0 >> 5;
  const int lane = (r & 15) | (((c0 >> 4) & 1) << 4);
  const int pos  = ((c0 >> 3) & 1) << 3;
  return (((msub * nksub + ksub) << 5) + lane) * 16 + pos;
}

__device__ __forceinline__ v16bf load_a_frag(const __bf16* frag_base, int sub) {
  const int lane = threadIdx.x & 31;
  return *(const v16bf*)(frag_base + ((sub << 5) + lane) * 16);
}

// B fragment from n-major tile [n][ldk]: 16 consecutive k at kh=16*(lane>=16).
__device__ __forceinline__ v16bf load_b_frag(const __bf16* tile, int n0, int ldk) {
  const int lane = threadIdx.x & 31;
  const int n  = n0 + (lane & 15);
  const int kh = (lane >> 4) << 4;
  return *(const v16bf*)(tile + n * ldk + kh);
}

__device__ __forceinline__ float sgnf(float x) {
  return (float)((x > 0.f) - (x < 0.f));
}

// ---- staging helpers (256 threads assumed) --------------------------------

// 128x32 fp32 tile -> fragment-major bf16 (two b128 LDS stores / thread)
__device__ __forceinline__ void stage_a_f32(__bf16* dst, const float* src_base,
                                            int ld, int tid) {
#pragma unroll
  for (int g = 0; g < 2; ++g) {
    const int e = tid * 16 + g * 8;
    const int r = e >> 5, c0 = e & 31;
    const float* src = src_base + (size_t)r * ld + c0;
    v8bf t;
#pragma unroll
    for (int i = 0; i < 8; ++i) t[i] = (__bf16)src[i];
    *(v8bf*)(dst + afrag_group(r, c0, 1)) = t;
  }
}

// 128x32 bf16 tile -> fragment-major bf16 (pure b128 copy)
__device__ __forceinline__ void stage_a_bf16(__bf16* dst, const __bf16* src_base,
                                             int ld, int tid) {
#pragma unroll
  for (int g = 0; g < 2; ++g) {
    const int e = tid * 16 + g * 8;
    const int r = e >> 5, c0 = e & 31;
    *(v8bf*)(dst + afrag_group(r, c0, 1)) =
        *(const v8bf*)(src_base + (size_t)r * ld + c0);
  }
}

// Transpose 32(k) x 64(n) fp32 -> n-major bf16 [n][32].
// Each thread: 4(k) x 2(n) patch = 4 coalesced b64 loads + 2 b64 LDS stores.
__device__ __forceinline__ void stage_b_f32t(__bf16* dst, const float* src_base,
                                             int ld, int tid) {
  const int k0 = (tid >> 5) * 4;
  const int nn = (tid & 31) * 2;
  const float* s = src_base + (size_t)k0 * ld + nn;
  const v2f r0 = *(const v2f*)(s);
  const v2f r1 = *(const v2f*)(s + ld);
  const v2f r2 = *(const v2f*)(s + 2 * (size_t)ld);
  const v2f r3 = *(const v2f*)(s + 3 * (size_t)ld);
  v4bf t0, t1;
  t0[0] = (__bf16)r0[0]; t0[1] = (__bf16)r1[0];
  t0[2] = (__bf16)r2[0]; t0[3] = (__bf16)r3[0];
  t1[0] = (__bf16)r0[1]; t1[1] = (__bf16)r1[1];
  t1[2] = (__bf16)r2[1]; t1[3] = (__bf16)r3[1];
  *(v4bf*)(dst + nn * 32 + k0)       = t0;
  *(v4bf*)(dst + (nn + 1) * 32 + k0) = t1;
}

// 64(d) x 32(k) bf16 plane slice, contiguous both sides (one b128 each way)
__device__ __forceinline__ void stage_v(__bf16* dst, const __bf16* src_base,
                                        int tid) {
  const int e = tid * 8, d = e >> 5, kc = e & 31;
  *(v8bf*)(dst + d * 32 + kc) = *(const v8bf*)(src_base + (size_t)d * SEQ + kc);
}

// ---------------------------------------------------------------------------
// Kernel 1: projection GEMM  out = x @ W + b   (M=4096, N=1024, K=1024)
// 128x64 tile, BK=32, double-buffered, 4 WMMA/wave/step.
// mode 0: out0[row*D+col] = bf16(v)                         (Q, K)
// mode 1: head-transposed planes [z][d][s]:
//         out0 = bf16(log(|v|+eps)), out1 = bf16(sign(v))   (V)
// ---------------------------------------------------------------------------
__global__ __launch_bounds__(256)
void proj_kernel(const float* __restrict__ X, const float* __restrict__ W,
                 const float* __restrict__ bias, __bf16* __restrict__ out0,
                 __bf16* __restrict__ out1, int mode) {
  __shared__ __align__(16) __bf16 As[2][128 * 32];
  __shared__ __align__(16) __bf16 Bs[2][64 * 32];
  const int tid  = threadIdx.x;
  const int wave = tid >> 5, lane = tid & 31;
  const int wm = wave >> 1, wn = wave & 1;
  const int m0 = blockIdx.y * 128;
  const int n0 = blockIdx.x * 64;
  const float* Xb = X + (size_t)m0 * D_MODEL;

  v8f acc[2][2] = {};

  stage_a_f32(As[0], Xb, D_MODEL, tid);
  stage_b_f32t(Bs[0], W + n0, D_MODEL, tid);

  const int NIT = D_MODEL / 32;
  for (int it = 0; it < NIT; ++it) {
    __syncthreads();
    const int buf = it & 1;
    if (it + 1 < NIT) {
      const int kt = (it + 1) * 32;
      stage_a_f32(As[buf ^ 1], Xb + kt, D_MODEL, tid);
      stage_b_f32t(Bs[buf ^ 1], W + (size_t)kt * D_MODEL + n0, D_MODEL, tid);
      if (it + 2 < NIT)
        __builtin_prefetch(Xb + (size_t)((tid * 16) >> 5) * D_MODEL + kt + 32, 0, 0);
    }
    v16bf b0 = load_b_frag(Bs[buf], wn * 32, 32);
    v16bf b1 = load_b_frag(Bs[buf], wn * 32 + 16, 32);
#pragma unroll
    for (int ms = 0; ms < 2; ++ms) {
      v16bf a = load_a_frag(As[buf], wm * 2 + ms);
      acc[ms][0] = wmma_bf16(a, b0, acc[ms][0]);
      acc[ms][1] = wmma_bf16(a, b1, acc[ms][1]);
    }
  }

#pragma unroll
  for (int ms = 0; ms < 2; ++ms) {
    const int mbase = m0 + (wm * 2 + ms) * 16 + ((lane >> 4) << 3);
#pragma unroll
    for (int ns = 0; ns < 2; ++ns) {
      const int col = n0 + wn * 32 + ns * 16 + (lane & 15);
      const float bb = bias[col];
#pragma unroll
      for (int r = 0; r < 8; ++r) {
        const int row = mbase + r;
        const float v = acc[ms][ns][r] + bb;
        if (mode == 0) {
          out0[(size_t)row * D_MODEL + col] = (__bf16)v;
        } else {
          // head-transposed plane: z = b*NHEADS + h, index [z][d][s]
          const size_t o = (size_t)(((row >> 11) << 4) | (col >> 6)) *
                               (DHEAD * SEQ) +
                           (size_t)(col & 63) * SEQ + (row & 2047);
          out0[o] = (__bf16)logf(fabsf(v) + EPSV);
          out1[o] = (__bf16)sgnf(v);
        }
      }
    }
  }
}

// ---------------------------------------------------------------------------
// Kernel 2: scores = (Q @ K^T) / sqrt(DH)  per (b,h). M=N=2048, K=64.
// 128x64 tile, whole K staged once, 8 WMMA/wave. Ks is naturally n-major.
// ---------------------------------------------------------------------------
__global__ __launch_bounds__(256)
void scores_kernel(const __bf16* __restrict__ qb, const __bf16* __restrict__ kb,
                   float* __restrict__ attn) {
  __shared__ __align__(16) __bf16 Qs[128 * 64];  // fragment-major, nksub=2
  __shared__ __align__(16) __bf16 Ks[64 * 64];   // [key][d]
  const int tid  = threadIdx.x;
  const int wave = tid >> 5, lane = tid & 31;
  const int wm = wave >> 1, wn = wave & 1;
  const int z = blockIdx.z;            // b*NHEADS + h
  const int bidx = z >> 4, h = z & 15;
  const int m0 = blockIdx.y * 128;     // query rows
  const int n0 = blockIdx.x * 64;      // key rows

  {  // Qs: 128x64, 4 groups of 8 / thread
#pragma unroll
    for (int g = 0; g < 4; ++g) {
      const int e = tid * 32 + g * 8;
      const int r = e >> 6, c0 = e & 63;
      *(v8bf*)(Qs + afrag_group(r, c0, 2)) = *(const v8bf*)(
          qb + (size_t)(bidx * SEQ + m0 + r) * D_MODEL + h * DHEAD + c0);
    }
    // Ks: 64x64, 2 groups of 8 / thread (natural n-major copy)
#pragma unroll
    for (int g = 0; g < 2; ++g) {
      const int e = tid * 16 + g * 8;
      const int r = e >> 6, c0 = e & 63;
      *(v8bf*)(Ks + r * 64 + c0) = *(const v8bf*)(
          kb + (size_t)(bidx * SEQ + n0 + r) * D_MODEL + h * DHEAD + c0);
    }
  }
  __syncthreads();

  v8f acc[2][2] = {};
#pragma unroll
  for (int ks = 0; ks < 2; ++ks) {
    v16bf b0 = load_b_frag(Ks + ks * 32, wn * 32, 64);
    v16bf b1 = load_b_frag(Ks + ks * 32, wn * 32 + 16, 64);
#pragma unroll
    for (int ms = 0; ms < 2; ++ms) {
      v16bf a = load_a_frag(Qs, (wm * 2 + ms) * 2 + ks);
      acc[ms][0] = wmma_bf16(a, b0, acc[ms][0]);
      acc[ms][1] = wmma_bf16(a, b1, acc[ms][1]);
    }
  }

  const float scale = 0.125f;  // 1/sqrt(64)
  float* ah = attn + (size_t)z * SEQ * SEQ;
#pragma unroll
  for (int ms = 0; ms < 2; ++ms) {
    const int mbase = m0 + (wm * 2 + ms) * 16 + ((lane >> 4) << 3);
    const int nbase = n0 + wn * 32 + (lane & 15);
#pragma unroll
    for (int r = 0; r < 8; ++r) {
      ah[(size_t)(mbase + r) * SEQ + nbase]      = acc[ms][0][r] * scale;
      ah[(size_t)(mbase + r) * SEQ + nbase + 16] = acc[ms][1][r] * scale;
    }
  }
}

// ---------------------------------------------------------------------------
// Kernel 3: row softmax over the key axis, in place. One block per row,
// float4 (b128) access: 8 consecutive floats per thread.
// ---------------------------------------------------------------------------
__global__ __launch_bounds__(256)
void softmax_kernel(float* __restrict__ attn) {
  float* p = attn + (size_t)blockIdx.x * SEQ;
  const int tid = threadIdx.x;
  __shared__ float red[256];

  v4f va = *(const v4f*)(p + tid * 8);
  v4f vb = *(const v4f*)(p + tid * 8 + 4);
  float mx = -3.4e38f;
#pragma unroll
  for (int i = 0; i < 4; ++i) mx = fmaxf(mx, fmaxf(va[i], vb[i]));
  red[tid] = mx;
  __syncthreads();
  for (int s = 128; s > 0; s >>= 1) {
    if (tid < s) red[tid] = fmaxf(red[tid], red[tid + s]);
    __syncthreads();
  }
  mx = red[0];
  __syncthreads();

  float sum = 0.f;
#pragma unroll
  for (int i = 0; i < 4; ++i) {
    va[i] = __expf(va[i] - mx); sum += va[i];
    vb[i] = __expf(vb[i] - mx); sum += vb[i];
  }
  red[tid] = sum;
  __syncthreads();
  for (int s = 128; s > 0; s >>= 1) {
    if (tid < s) red[tid] += red[tid + s];
    __syncthreads();
  }
  const float inv = 1.f / red[0];
  *(v4f*)(p + tid * 8)     = va * inv;
  *(v4f*)(p + tid * 8 + 4) = vb * inv;
}

// ---------------------------------------------------------------------------
// Kernel 4: dual AV GEMM per (b,h): M=2048, N=64, K=2048, double-buffered.
//   lo = attn @ v_log ; sw = attn @ v_sign ; ctx = exp(lo) * sign(sw + eps)
// V planes head-transposed [z][d][s] so B staging is b128 on both sides.
// 8 WMMA/wave/step.
// ---------------------------------------------------------------------------
__global__ __launch_bounds__(256)
void av_kernel(const float* __restrict__ attn, const __bf16* __restrict__ vlT,
               const __bf16* __restrict__ vsT, __bf16* __restrict__ ctx) {
  __shared__ __align__(16) __bf16 As[2][128 * 32];
  __shared__ __align__(16) __bf16 Vl[2][64 * 32];
  __shared__ __align__(16) __bf16 Vs[2][64 * 32];
  const int tid  = threadIdx.x;
  const int wave = tid >> 5, lane = tid & 31;
  const int wm = wave >> 1, wn = wave & 1;
  const int z = blockIdx.y;            // b*NHEADS + h
  const int bidx = z >> 4, h = z & 15;
  const int m0 = blockIdx.x * 128;
  const float*  ah  = attn + (size_t)z * SEQ * SEQ + (size_t)m0 * SEQ;
  const __bf16* vlz = vlT + (size_t)z * DHEAD * SEQ;
  const __bf16* vsz = vsT + (size_t)z * DHEAD * SEQ;

  v8f cl[2][2] = {}, cs[2][2] = {};

  stage_a_f32(As[0], ah, SEQ, tid);
  stage_v(Vl[0], vlz, tid);
  stage_v(Vs[0], vsz, tid);

  const int NIT = SEQ / 32;
  for (int it = 0; it < NIT; ++it) {
    __syncthreads();
    const int buf = it & 1;
    if (it + 1 < NIT) {
      const int kt = (it + 1) * 32;
      stage_a_f32(As[buf ^ 1], ah + kt, SEQ, tid);
      stage_v(Vl[buf ^ 1], vlz + kt, tid);
      stage_v(Vs[buf ^ 1], vsz + kt, tid);
      if (it + 2 < NIT)
        __builtin_prefetch(ah + (size_t)((tid * 16) >> 5) * SEQ + kt + 32, 0, 0);
    }
    v16bf bl0 = load_b_frag(Vl[buf], wn * 32, 32);
    v16bf bl1 = load_b_frag(Vl[buf], wn * 32 + 16, 32);
    v16bf bs0 = load_b_frag(Vs[buf], wn * 32, 32);
    v16bf bs1 = load_b_frag(Vs[buf], wn * 32 + 16, 32);
#pragma unroll
    for (int ms = 0; ms < 2; ++ms) {
      v16bf a = load_a_frag(As[buf], wm * 2 + ms);
      cl[ms][0] = wmma_bf16(a, bl0, cl[ms][0]);
      cl[ms][1] = wmma_bf16(a, bl1, cl[ms][1]);
      cs[ms][0] = wmma_bf16(a, bs0, cs[ms][0]);
      cs[ms][1] = wmma_bf16(a, bs1, cs[ms][1]);
    }
  }

#pragma unroll
  for (int ms = 0; ms < 2; ++ms) {
    const int mbase = m0 + (wm * 2 + ms) * 16 + ((lane >> 4) << 3);
#pragma unroll
    for (int ns = 0; ns < 2; ++ns) {
      const int d = wn * 32 + ns * 16 + (lane & 15);
#pragma unroll
      for (int r = 0; r < 8; ++r) {
        const int row = mbase + r;
        const float o = __expf(cl[ms][ns][r]) * sgnf(cs[ms][ns][r] + EPSV);
        ctx[(size_t)(bidx * SEQ + row) * D_MODEL + h * DHEAD + d] = (__bf16)o;
      }
    }
  }
}

// ---------------------------------------------------------------------------
// Kernel 5: output projection  out = ctx @ W_o + b_o  (fp32 out),
// double-buffered; A staging is pure b128 copy.
// ---------------------------------------------------------------------------
__global__ __launch_bounds__(256)
void oproj_kernel(const __bf16* __restrict__ Ctx, const float* __restrict__ W,
                  const float* __restrict__ bias, float* __restrict__ out) {
  __shared__ __align__(16) __bf16 As[2][128 * 32];
  __shared__ __align__(16) __bf16 Bs[2][64 * 32];
  const int tid  = threadIdx.x;
  const int wave = tid >> 5, lane = tid & 31;
  const int wm = wave >> 1, wn = wave & 1;
  const int m0 = blockIdx.y * 128;
  const int n0 = blockIdx.x * 64;
  const __bf16* Cb = Ctx + (size_t)m0 * D_MODEL;

  v8f acc[2][2] = {};

  stage_a_bf16(As[0], Cb, D_MODEL, tid);
  stage_b_f32t(Bs[0], W + n0, D_MODEL, tid);

  const int NIT = D_MODEL / 32;
  for (int it = 0; it < NIT; ++it) {
    __syncthreads();
    const int buf = it & 1;
    if (it + 1 < NIT) {
      const int kt = (it + 1) * 32;
      stage_a_bf16(As[buf ^ 1], Cb + kt, D_MODEL, tid);
      stage_b_f32t(Bs[buf ^ 1], W + (size_t)kt * D_MODEL + n0, D_MODEL, tid);
    }
    v16bf b0 = load_b_frag(Bs[buf], wn * 32, 32);
    v16bf b1 = load_b_frag(Bs[buf], wn * 32 + 16, 32);
#pragma unroll
    for (int ms = 0; ms < 2; ++ms) {
      v16bf a = load_a_frag(As[buf], wm * 2 + ms);
      acc[ms][0] = wmma_bf16(a, b0, acc[ms][0]);
      acc[ms][1] = wmma_bf16(a, b1, acc[ms][1]);
    }
  }

#pragma unroll
  for (int ms = 0; ms < 2; ++ms) {
    const int mbase = m0 + (wm * 2 + ms) * 16 + ((lane >> 4) << 3);
#pragma unroll
    for (int ns = 0; ns < 2; ++ns) {
      const int col = n0 + wn * 32 + ns * 16 + (lane & 15);
      const float bb = bias[col];
#pragma unroll
      for (int r = 0; r < 8; ++r)
        out[(size_t)(mbase + r) * D_MODEL + col] = acc[ms][ns][r] + bb;
    }
  }
}

// ---------------------------------------------------------------------------
// Launcher. d_out = [out (B,S,D) fp32 | attn (B,H,S,S) fp32].
// Workspace (bf16): Q | K | log|V| (head-T) | sign(V) (head-T) | ctx = 40 MB.
// ---------------------------------------------------------------------------
extern "C" void kernel_launch(void* const* d_in, const int* in_sizes, int n_in,
                              void* d_out, int out_size, void* d_ws, size_t ws_size,
                              hipStream_t stream) {
  const float* x  = (const float*)d_in[0];
  const float* wq = (const float*)d_in[1];
  const float* bq = (const float*)d_in[2];
  const float* wk = (const float*)d_in[3];
  const float* bk = (const float*)d_in[4];
  const float* wv = (const float*)d_in[5];
  const float* bv = (const float*)d_in[6];
  const float* wo = (const float*)d_in[7];
  const float* bo = (const float*)d_in[8];

  float* out  = (float*)d_out;
  float* attn = out + (size_t)MTOT * D_MODEL;

  const size_t tensz = (size_t)MTOT * D_MODEL;  // elements per bf16 plane
  __bf16* qb  = (__bf16*)d_ws;
  __bf16* kb  = qb + tensz;
  __bf16* vlT = kb + tensz;   // [z][d][s]
  __bf16* vsT = vlT + tensz;  // [z][d][s]
  __bf16* ctx = vsT + tensz;

  const dim3 blk(256);
  const dim3 gproj(D_MODEL / 64, MTOT / 128);       // (16, 32)
  proj_kernel<<<gproj, blk, 0, stream>>>(x, wq, bq, qb, nullptr, 0);
  proj_kernel<<<gproj, blk, 0, stream>>>(x, wk, bk, kb, nullptr, 0);
  proj_kernel<<<gproj, blk, 0, stream>>>(x, wv, bv, vlT, vsT, 1);

  const dim3 gsc(SEQ / 64, SEQ / 128, NBATCH * NHEADS);  // (32, 16, 32)
  scores_kernel<<<gsc, blk, 0, stream>>>(qb, kb, attn);

  softmax_kernel<<<dim3(NBATCH * NHEADS * SEQ), blk, 0, stream>>>(attn);

  av_kernel<<<dim3(SEQ / 128, NBATCH * NHEADS), blk, 0, stream>>>(attn, vlT, vsT, ctx);

  oproj_kernel<<<gproj, blk, 0, stream>>>(ctx, wo, bo, out);
}
